// MultiHeadSelfAttention_11149735100445
// MI455X (gfx1250) — matrix operand order repaired and checked by reading, not compiled, not run on hardware
//
#include <hip/hip_runtime.h>
#include <hip/hip_bf16.h>

#define BB  4
#define SS  2048
#define DD  1024
#define HH  16
#define DKK 64

typedef __attribute__((ext_vector_type(16))) __bf16 v16bf;
typedef __attribute__((ext_vector_type(8)))  __bf16 v8bf;
typedef __attribute__((ext_vector_type(8)))  float  v8f;

// fp32 -> bf16, round-to-nearest-even, via bit ops
static __device__ __forceinline__ __bf16 f2b(float f) {
  unsigned int u = __builtin_bit_cast(unsigned int, f);
  unsigned int r = (u + 0x7FFFu + ((u >> 16) & 1u)) >> 16;
  unsigned short s = (unsigned short)r;
  return __builtin_bit_cast(__bf16, s);
}

// ---------------- fp32 -> bf16 convert (8 elems/thread) ----------------
__global__ __launch_bounds__(256) void cvt_f32_bf16(const float* __restrict__ in,
                                                    __bf16* __restrict__ out, int n) {
  int i = (blockIdx.x * 256 + threadIdx.x) * 8;
  if (i + 8 <= n) {
    float4 a = *(const float4*)(in + i);
    float4 b = *(const float4*)(in + i + 4);
    v8bf o;
    o[0] = f2b(a.x); o[1] = f2b(a.y); o[2] = f2b(a.z); o[3] = f2b(a.w);
    o[4] = f2b(b.x); o[5] = f2b(b.y); o[6] = f2b(b.z); o[7] = f2b(b.w);
    *(v8bf*)(out + i) = o;
  }
}

// ---------------- bf16 WMMA GEMM: Y = preScale * (X[M,K] * W[N,K]^T) ----------------
// MODE 0: Yb = bf16 in [B,H,S,DK]   (Q and K; preScale folds 1/sqrt(DK) into Q)
// MODE 1: Yb = bf16 in [B,H,DK,S]   (V transposed -> PV B-operand contiguous)
// MODE 3: Yf = f32  in [M,N] + bias (final projection)
// wave tile: 16 rows x 128 cols (8 accumulators); 8 waves/block -> 128x128 block tile
template<int MODE>
__global__ __launch_bounds__(256) void gemm_bf16_wmma(
    const __bf16* __restrict__ X, const __bf16* __restrict__ W,
    __bf16* __restrict__ Yb, float* __restrict__ Yf, const float* __restrict__ bias,
    float preScale) {
  const int lane = threadIdx.x & 31;
  const int wave = threadIdx.x >> 5;
  const int lm = lane & 15;   // A: row-in-tile, B/C: col-in-tile
  const int lh = lane >> 4;   // half-wave select
  const int rowBase = blockIdx.y * 128 + wave * 16;
  const int colBase = blockIdx.x * 128;

  v8f acc[8] = {};

  const __bf16* xrow = X + (size_t)(rowBase + lm) * DD;
  for (int k0 = 0; k0 < DD; k0 += 32) {
    if (k0 + 64 < DD) __builtin_prefetch((const void*)(xrow + k0 + 64), 0, 3);
    // A operand (16x32 bf16): lane lm = row; K contiguous [lh*16, lh*16+16)
    v8bf alo = *(const v8bf*)(xrow + k0 + lh * 16);
    v8bf ahi = *(const v8bf*)(xrow + k0 + lh * 16 + 8);
    v16bf a;
#pragma unroll
    for (int i = 0; i < 8; ++i) { a[i] = alo[i]; a[8 + i] = ahi[i]; }
    // B operands: lane lm = output col n; K packed contiguous (row of W)
#pragma unroll
    for (int t = 0; t < 8; ++t) {
      const __bf16* wp = W + (size_t)(colBase + t * 16 + lm) * DD + k0 + lh * 16;
      v16bf bm = *(const v16bf*)wp;
      acc[t] = __builtin_amdgcn_wmma_f32_16x16x32_bf16(false, a, false, bm,
                                                       (short)0, acc[t], false, false);
    }
  }

#pragma unroll
  for (int t = 0; t < 8; ++t) {
    const int n = colBase + t * 16 + lm;
#pragma unroll
    for (int i = 0; i < 8; ++i) {
      const int row = rowBase + i + 8 * lh;   // C layout: M = i + 8*(lane/16)
      float v = acc[t][i];
      if (MODE == 3) {
        Yf[(size_t)row * DD + n] = v + bias[n];
      } else {
        v *= preScale;
        const int b = row / SS, s = row % SS;
        const int h = n / DKK, dk = n % DKK;
        if (MODE == 0)
          Yb[(((size_t)(b * HH + h)) * SS + s) * DKK + dk] = f2b(v);
        else
          Yb[(((size_t)(b * HH + h)) * DKK + dk) * SS + s] = f2b(v);
      }
    }
  }
}

// ---------------- causal flash attention: 1 wave per (b,h,16 queries) ----------------
// Row max via shfl_xor butterfly over the 16 lanes sharing a row (C layout);
// row SUM via one extra WMMA against an all-ones B matrix (layout-invariant),
// landing the per-row sums directly in each lane's C-layout elements.
// Running m/l stats live replicated in registers; LDS only re-layouts P (C->A).
__global__ __launch_bounds__(32) void flash_attn_wmma(
    const __bf16* __restrict__ Q, const __bf16* __restrict__ K,
    const __bf16* __restrict__ Vt, __bf16* __restrict__ Out) {
  __shared__ alignas(16) __bf16 sP[16][32];

  const int lane = threadIdx.x;
  const int lm = lane & 15, lh = lane >> 4;

  const int qb = blockIdx.x & (SS / 16 - 1);
  const int h  = (blockIdx.x / (SS / 16)) & (HH - 1);
  const int b  = blockIdx.x / ((SS / 16) * HH);
  const int qbase = qb * 16;

  const __bf16* Qp = Q  + (size_t)(b * HH + h) * SS * DKK;
  const __bf16* Kp = K  + (size_t)(b * HH + h) * SS * DKK;
  const __bf16* Vp = Vt + (size_t)(b * HH + h) * DKK * SS;

  // Q A-operands (Q already carries the 1/sqrt(DK) scale): two 16x32 chunks
  v16bf aQ[2];
#pragma unroll
  for (int c = 0; c < 2; ++c) {
    const __bf16* qp = Qp + (size_t)(qbase + lm) * DKK + c * 32 + lh * 16;
    v8bf lo = *(const v8bf*)qp;
    v8bf hi = *(const v8bf*)(qp + 8);
#pragma unroll
    for (int i = 0; i < 8; ++i) { aQ[c][i] = lo[i]; aQ[c][8 + i] = hi[i]; }
  }

  // all-ones bf16 B operand for row-sum WMMA
  const __bf16 one_bf = __builtin_bit_cast(__bf16, (unsigned short)0x3F80);
  v16bf ones;
#pragma unroll
  for (int i = 0; i < 16; ++i) ones[i] = one_bf;

  float mrun[8], lrun[8];
#pragma unroll
  for (int i = 0; i < 8; ++i) { mrun[i] = -1e30f; lrun[i] = 0.f; }
  v8f o[4] = {};

  const int nkb = (qbase + 47) / 32;        // causal: key blocks of 32
  for (int jb = 0; jb < nkb; ++jb) {
    const int kbase = jb * 32;
    if (jb + 1 < nkb) {
      __builtin_prefetch((const void*)(Kp + (size_t)(kbase + 32 + lane) * DKK), 0, 3);
      __builtin_prefetch((const void*)(Vp + (size_t)(lane * 2) * SS + kbase + 32), 0, 3);
    }

    // ---- scores: two 16x16 tiles, K-dim = DK=64 in two chunks ----
    v8f sc[2];
#pragma unroll
    for (int t = 0; t < 2; ++t) {
      v8f s = {};
#pragma unroll
      for (int c = 0; c < 2; ++c) {
        const __bf16* kp = Kp + (size_t)(kbase + t * 16 + lm) * DKK + c * 32 + lh * 16;
        v16bf bK = *(const v16bf*)kp;
        s = __builtin_amdgcn_wmma_f32_16x16x32_bf16(false, aQ[c], false, bK,
                                                    (short)0, s, false, false);
      }
      sc[t] = s;
    }

    // ---- causal mask only on the (single) partial diagonal block ----
    if (kbase + 31 > qbase) {               // wave-uniform branch
#pragma unroll
      for (int t = 0; t < 2; ++t) {
        const int kg = kbase + t * 16 + lm;
#pragma unroll
        for (int i = 0; i < 8; ++i) {
          const int q = qbase + i + 8 * lh;
          if (kg > q) sc[t][i] = -1e30f;
        }
      }
    }

    // ---- row max via 16-lane butterfly (rows live in 16-lane groups) ----
    v8f mx;
#pragma unroll
    for (int i = 0; i < 8; ++i) mx[i] = fmaxf(sc[0][i], sc[1][i]);
#pragma unroll
    for (int d = 1; d < 16; d <<= 1)
#pragma unroll
      for (int i = 0; i < 8; ++i) mx[i] = fmaxf(mx[i], __shfl_xor(mx[i], d, 32));

    float scale[8];
#pragma unroll
    for (int i = 0; i < 8; ++i) {
      float mnew = fmaxf(mrun[i], mx[i]);
      scale[i] = __expf(mrun[i] - mnew);
      mrun[i] = mnew;
    }

    // ---- p = exp(s - m); stage P through LDS (C-layout write -> A-layout read) ----
#pragma unroll
    for (int i = 0; i < 8; ++i) {
      const int m = i + 8 * lh;
      sP[m][lm]      = f2b(__expf(sc[0][i] - mrun[i]));
      sP[m][16 + lm] = f2b(__expf(sc[1][i] - mrun[i]));
    }
    __syncthreads();

    // ---- rescale O accumulators ----
#pragma unroll
    for (int t = 0; t < 4; ++t)
#pragma unroll
      for (int i = 0; i < 8; ++i) o[t][i] *= scale[i];

    v16bf aP;
    {
      const __bf16* pp = &sP[lm][lh * 16];
      v8bf lo = *(const v8bf*)pp;
      v8bf hi = *(const v8bf*)(pp + 8);
#pragma unroll
      for (int i = 0; i < 8; ++i) { aP[i] = lo[i]; aP[8 + i] = hi[i]; }
    }

    // ---- row sums in one WMMA: P(16x32) * ones(32x16) -> C[m][*] = sum_k P[m][k] ----
    v8f rs = __builtin_amdgcn_wmma_f32_16x16x32_bf16(false, aP, false, ones,
                                                     (short)0, v8f{}, false, false);
#pragma unroll
    for (int i = 0; i < 8; ++i) lrun[i] = lrun[i] * scale[i] + rs[i];

    // ---- O += P(16x32) * V(32x64); V^T layout -> contiguous B loads ----
#pragma unroll
    for (int t = 0; t < 4; ++t) {
      const __bf16* vp = Vp + (size_t)(t * 16 + lm) * SS + kbase + lh * 16;
      v16bf bV = *(const v16bf*)vp;
      o[t] = __builtin_amdgcn_wmma_f32_16x16x32_bf16(false, aP, false, bV,
                                                     (short)0, o[t], false, false);
    }
    __syncthreads();
  }

  // ---- normalize and store to [B,S,D] bf16 ----
  float inv[8];
#pragma unroll
  for (int i = 0; i < 8; ++i) inv[i] = 1.f / lrun[i];
#pragma unroll
  for (int t = 0; t < 4; ++t) {
    const int dk = t * 16 + lm;
#pragma unroll
    for (int i = 0; i < 8; ++i) {
      const int q = qbase + i + 8 * lh;
      Out[((size_t)b * SS + q) * DD + h * DKK + dk] = f2b(o[t][i] * inv[i]);
    }
  }
}

extern "C" void kernel_launch(void* const* d_in, const int* in_sizes, int n_in,
                              void* d_out, int out_size, void* d_ws, size_t ws_size,
                              hipStream_t stream) {
  const float* x  = (const float*)d_in[0];
  const float* wq = (const float*)d_in[1];
  const float* wk = (const float*)d_in[2];
  const float* wv = (const float*)d_in[3];
  const float* wo = (const float*)d_in[4];
  const float* bo = (const float*)d_in[5];
  float* out = (float*)d_out;

  const size_t xN = (size_t)BB * SS * DD;   // 8,388,608
  const size_t wN = (size_t)DD * DD;        // 1,048,576

  char* ws = (char*)d_ws;
  __bf16* xb  = (__bf16*)ws; ws += xN * 2;
  __bf16* wqb = (__bf16*)ws; ws += wN * 2;
  __bf16* wkb = (__bf16*)ws; ws += wN * 2;
  __bf16* wvb = (__bf16*)ws; ws += wN * 2;
  __bf16* wob = (__bf16*)ws; ws += wN * 2;
  __bf16* qB  = (__bf16*)ws; ws += xN * 2;
  __bf16* kB  = (__bf16*)ws; ws += xN * 2;
  __bf16* vtB = (__bf16*)ws; ws += xN * 2;
  __bf16* aB  = (__bf16*)ws; ws += xN * 2;   // ~88 MB total

  cvt_f32_bf16<<<(int)(xN / (256 * 8)), 256, 0, stream>>>(x,  xb,  (int)xN);
  cvt_f32_bf16<<<(int)(wN / (256 * 8)), 256, 0, stream>>>(wq, wqb, (int)wN);
  cvt_f32_bf16<<<(int)(wN / (256 * 8)), 256, 0, stream>>>(wk, wkb, (int)wN);
  cvt_f32_bf16<<<(int)(wN / (256 * 8)), 256, 0, stream>>>(wv, wvb, (int)wN);
  cvt_f32_bf16<<<(int)(wN / (256 * 8)), 256, 0, stream>>>(wo, wob, (int)wN);

  dim3 gg(DD / 128, (BB * SS) / 128);
  const float qscale = 0.125f;              // 1/sqrt(64) folded into Q
  gemm_bf16_wmma<0><<<gg, 256, 0, stream>>>(xb, wqb, qB,  nullptr, nullptr, qscale);
  gemm_bf16_wmma<0><<<gg, 256, 0, stream>>>(xb, wkb, kB,  nullptr, nullptr, 1.0f);
  gemm_bf16_wmma<1><<<gg, 256, 0, stream>>>(xb, wvb, vtB, nullptr, nullptr, 1.0f);

  flash_attn_wmma<<<BB * HH * (SS / 16), 32, 0, stream>>>(qB, kB, vtB, aB);

  gemm_bf16_wmma<3><<<gg, 256, 0, stream>>>(aB, wob, nullptr, out, bo, 1.0f);
}
